// EdgeNetwork_40037685133515
// MI455X (gfx1250) — matrix operand order repaired and checked by reading, not compiled, not run on hardware
//
#include <hip/hip_runtime.h>

typedef _Float16 v16h __attribute__((ext_vector_type(16)));
typedef _Float16 v8h  __attribute__((ext_vector_type(8)));
typedef float    v8f  __attribute__((ext_vector_type(8)));
typedef float    v4f  __attribute__((ext_vector_type(4)));

#define BONDK   64
#define NCOLS   1024
#define KHALFS  ((BONDK + 1) * NCOLS)              // 66560 f16 (row 64 = bias)
#define ESUB    4                                   // 4 sub-tiles of 16 edges
#define EPT     (ESUB * 16)                         // 64 edges per iteration
#define BHALFS  (EPT * BONDK)                       // per-wave bond staging: 4096 f16
#define NTHREADS 512                                // 16 waves: one block owns the WGP
#define NWAVES   (NTHREADS / 32)
#define LDS_BYTES ((KHALFS + NWAVES * BHALFS) * 2)  // 264,192 B of 320 KB WGP LDS
#define NBLOCKS  160

__device__ __forceinline__ v16h cat8h(v8h lo, v8h hi) {
  return __builtin_shufflevector(lo, hi, 0,1,2,3,4,5,6,7,8,9,10,11,12,13,14,15);
}

// out[e,i] = sum_c bond[e,c]*(nbr[e,:].K[c,i*32:+32]) + nbr[e,:].bias[i*32:+32]
// per k-chunk c: D += A_c @ B_c, A_c = diag(bond[:,c])*nbr_tile, B_c[j,i] = K[c,i*32+j]
__global__ void __launch_bounds__(NTHREADS, 1)
edge_network_fused(const float* __restrict__ atom,       // [N_ATOMS, 32]
                   const float* __restrict__ bond,       // [N_EDGES, 64]
                   const long long* __restrict__ pairs,  // [N_EDGES, 2] int64
                   const float* __restrict__ kmat,       // [64, 1024]
                   const float* __restrict__ bias,       // [1024]
                   float* __restrict__ out,              // [N_ATOMS, 32] (pre-zeroed)
                   int n_edges)
{
  extern __shared__ _Float16 lds[];
  _Float16* kimg = lds;                                  // [65][1024] f16 K + bias

  for (int idx = threadIdx.x; idx < KHALFS; idx += blockDim.x) {
    float v = (idx < BONDK * NCOLS) ? kmat[idx] : bias[idx - BONDK * NCOLS];
    kimg[idx] = (_Float16)v;
  }
  __syncthreads();

  const int lane = threadIdx.x & 31;
  const int wid  = threadIdx.x >> 5;
  _Float16* bnd  = lds + KHALFS + wid * BHALFS;          // wave-private [64 e][64 c] f16

  const int gwave = blockIdx.x * NWAVES + wid;
  const int nwave = gridDim.x * NWAVES;
  const int half  = lane >> 4;
  const int erow  = lane & 15;
  const int jA    = half ? 16 : 0;   // A-operand K base (contiguous k = 16h + s)
  const int jB    = half ? 16 : 0;   // B-operand K base (contiguous k = 16h + s, same as A)

  const int ntiles = n_edges / EPT;
  for (int t = gwave; t < ntiles; t += nwave) {
    const int e0 = t * EPT;

    // --- Stage this wave's 64x64 bond tile to LDS as f16.
    // Per-wave LDS ops are in-order: no barrier needed before the reads below.
    #pragma unroll
    for (int p = 0; p < 2; ++p) {
      const int er = p * 32 + lane;
      const float* bp = bond + (size_t)(e0 + er) * 64;
      _Float16* dst = bnd + er * 64;
      #pragma unroll
      for (int g = 0; g < 4; ++g) {
        v4f b0 = __builtin_nontemporal_load((const v4f*)(bp + g * 16));
        v4f b1 = __builtin_nontemporal_load((const v4f*)(bp + g * 16 + 4));
        v4f b2 = __builtin_nontemporal_load((const v4f*)(bp + g * 16 + 8));
        v4f b3 = __builtin_nontemporal_load((const v4f*)(bp + g * 16 + 12));
        v8h h0, h1;
        #pragma unroll
        for (int k = 0; k < 4; ++k) {
          h0[k] = (_Float16)b0[k]; h0[4 + k] = (_Float16)b1[k];
          h1[k] = (_Float16)b2[k]; h1[4 + k] = (_Float16)b3[k];
        }
        *(v8h*)(dst + g * 16)     = h0;
        *(v8h*)(dst + g * 16 + 8) = h1;
      }
    }

    // --- Gather neighbor rows into A-operand layout (f16). Atom table is
    // 6.4 MB -> L2-resident; keep these cached (no NT hint).
    v16h anbr[ESUB];
    #pragma unroll
    for (int s = 0; s < ESUB; ++s) {
      const int e = e0 + s * 16 + erow;
      const long long nb = pairs[2 * e + 1];
      const float* ap = atom + (size_t)nb * 32 + jA;
      v4f q0 = *(const v4f*)(ap);
      v4f q1 = *(const v4f*)(ap + 4);
      v4f q2 = *(const v4f*)(ap + 8);
      v4f q3 = *(const v4f*)(ap + 12);
      v16h a;
      #pragma unroll
      for (int k = 0; k < 4; ++k) {
        a[k]      = (_Float16)q0[k];
        a[4 + k]  = (_Float16)q1[k];
        a[8 + k]  = (_Float16)q2[k];
        a[12 + k] = (_Float16)q3[k];
      }
      anbr[s] = a;
    }

    v8f acc[2 * ESUB];
    #pragma unroll
    for (int x = 0; x < 2 * ESUB; ++x) {
      #pragma unroll
      for (int r = 0; r < 8; ++r) acc[x][r] = 0.0f;
    }

    // --- Main k-chunk loop: dynamic (no spills), unroll 2 so the scheduler
    // double-buffers scaled-A registers and prefetches next-chunk B tiles.
    #pragma unroll 2
    for (int c = 0; c < BONDK; ++c) {
      const _Float16* kc = kimg + c * NCOLS;
      v16h B0 = cat8h(*(const v8h*)(kc + erow * 32 + jB),
                      *(const v8h*)(kc + erow * 32 + jB + 8));
      v16h B1 = cat8h(*(const v8h*)(kc + (erow + 16) * 32 + jB),
                      *(const v8h*)(kc + (erow + 16) * 32 + jB + 8));
      #pragma unroll
      for (int s = 0; s < ESUB; ++s) {
        const _Float16 sc = bnd[(s * 16 + erow) * 64 + c];   // ds_load_u16
        v16h as = anbr[s] * sc;                              // v_pk_mul_f16 splat
        acc[2*s+0] = __builtin_amdgcn_wmma_f32_16x16x32_f16(false, as, false, B0,
                                                            (short)0, acc[2*s+0], false, false);
        acc[2*s+1] = __builtin_amdgcn_wmma_f32_16x16x32_f16(false, as, false, B1,
                                                            (short)0, acc[2*s+1], false, false);
      }
    }

    // --- Bias chunk (scale = 1).
    {
      const _Float16* kc = kimg + BONDK * NCOLS;
      v16h B0 = cat8h(*(const v8h*)(kc + erow * 32 + jB),
                      *(const v8h*)(kc + erow * 32 + jB + 8));
      v16h B1 = cat8h(*(const v8h*)(kc + (erow + 16) * 32 + jB),
                      *(const v8h*)(kc + (erow + 16) * 32 + jB + 8));
      #pragma unroll
      for (int s = 0; s < ESUB; ++s) {
        acc[2*s+0] = __builtin_amdgcn_wmma_f32_16x16x32_f16(false, anbr[s], false, B0,
                                                            (short)0, acc[2*s+0], false, false);
        acc[2*s+1] = __builtin_amdgcn_wmma_f32_16x16x32_f16(false, anbr[s], false, B1,
                                                            (short)0, acc[2*s+1], false, false);
      }
    }

    // --- Scatter: C-layout row M = r + 8*half, col N = erow (+16 for i-tile 1).
    #pragma unroll
    for (int s = 0; s < ESUB; ++s) {
      #pragma unroll
      for (int r = 0; r < 8; ++r) {
        const int e   = e0 + s * 16 + r + half * 8;
        const int seg = (int)pairs[2 * e];
        float* op = out + (size_t)seg * 32 + erow;
        __hip_atomic_fetch_add(op,      acc[2*s+0][r], __ATOMIC_RELAXED, __HIP_MEMORY_SCOPE_AGENT);
        __hip_atomic_fetch_add(op + 16, acc[2*s+1][r], __ATOMIC_RELAXED, __HIP_MEMORY_SCOPE_AGENT);
      }
    }
  }
}

extern "C" void kernel_launch(void* const* d_in, const int* in_sizes, int n_in,
                              void* d_out, int out_size, void* d_ws, size_t ws_size,
                              hipStream_t stream) {
  const float*     atom  = (const float*)d_in[0];
  const float*     bond  = (const float*)d_in[1];
  const long long* pairs = (const long long*)d_in[2];
  const float*     kmat  = (const float*)d_in[3];
  const float*     bias  = (const float*)d_in[4];
  float*           out   = (float*)d_out;

  const int n_edges = in_sizes[2] / 2;  // 320000 (divisible by 64)

  hipMemsetAsync(out, 0, (size_t)out_size * sizeof(float), stream);

  hipFuncSetAttribute(reinterpret_cast<const void*>(edge_network_fused),
                      hipFuncAttributeMaxDynamicSharedMemorySize, LDS_BYTES);

  edge_network_fused<<<NBLOCKS, NTHREADS, LDS_BYTES, stream>>>(atom, bond, pairs, kmat,
                                                               bias, out, n_edges);
}
